// GatedMLPMOE_88776974008633
// MI455X (gfx1250) — compile-verified, matrix operand order checked by reading, not run on hardware
//
#include <hip/hip_runtime.h>
#include <math.h>

#define NUM_EXPERTS 8
#define HIDDEN 2048
#define INTER 4096
#define TOKENS 1024
#define NSLOTS (TOKENS * 2)

typedef __bf16 bf16_t;
typedef bf16_t v16bf __attribute__((ext_vector_type(16)));
typedef bf16_t v8bf  __attribute__((ext_vector_type(8)));
typedef float  v8f   __attribute__((ext_vector_type(8)));

// ---------------------------------------------------------------------------
// A-fragment (16-bit 16x32, ISA 7.12.2): lane = M (mod 16), g = lane>>4.
// Lane halves j=0..7 hold K = 8g..8g+7 ; j=8..15 hold K = 16+8g..16+8g+7.
// Source is bf16 row-major, so each lane does two contiguous 16B loads.
// ---------------------------------------------------------------------------
static __device__ __forceinline__ v16bf load_a_frag(const bf16_t* __restrict__ row, int g) {
    v8bf lo = *reinterpret_cast<const v8bf*>(row + 8 * g);
    v8bf hi = *reinterpret_cast<const v8bf*>(row + 16 + 8 * g);
    v16bf a;
#pragma unroll
    for (int j = 0; j < 8; ++j) { a[j] = lo[j]; a[j + 8] = hi[j]; }
    return a;
}

// ---------------------------------------------------------------------------
// B-fragment (32x16), mirrored layout: lane = N (mod 16), same K split.
// Weight rows (output index) are contiguous in K in W13/W2, so per-lane we
// stream two contiguous 32B fp32 runs and convert to bf16 in registers.
// ---------------------------------------------------------------------------
static __device__ __forceinline__ v16bf load_b_frag_f32(const float* __restrict__ row, int g) {
    const float* p0 = row + 8 * g;
    const float* p1 = row + 16 + 8 * g;
    v16bf b;
#pragma unroll
    for (int j = 0; j < 8; ++j) { b[j] = (bf16_t)p0[j]; b[j + 8] = (bf16_t)p1[j]; }
    return b;
}

static __device__ __forceinline__ v8f wmma_bf16(v16bf a, v16bf b, v8f c) {
    return __builtin_amdgcn_wmma_f32_16x16x32_bf16(false, a, false, b, (short)0, c, false, false);
}

// ---------------------------------------------------------------------------
// Router: softmax -> top2 (strict-> scan == jax tie-break) -> renormalize
// (== 2-way softmax of the two winning logits). Compacts (token, expert)
// pairs into per-expert contiguous slot ranges.
// ---------------------------------------------------------------------------
__global__ void __launch_bounds__(TOKENS)
moe_router(const float* __restrict__ logits, int* __restrict__ counts,
           int* __restrict__ bases, int* __restrict__ tok_of_slot,
           int* __restrict__ slots, float* __restrict__ wts) {
    __shared__ int lcnt[NUM_EXPERTS];
    __shared__ int lbase[NUM_EXPERTS];
    const int t = threadIdx.x;
    if (t < NUM_EXPERTS) lcnt[t] = 0;
    __syncthreads();

    float l[NUM_EXPERTS];
#pragma unroll
    for (int e = 0; e < NUM_EXPERTS; ++e) l[e] = logits[t * NUM_EXPERTS + e];
    int e0 = 0;
#pragma unroll
    for (int e = 1; e < NUM_EXPERTS; ++e) if (l[e] > l[e0]) e0 = e;
    int e1 = -1;
#pragma unroll
    for (int e = 0; e < NUM_EXPERTS; ++e)
        if (e != e0 && (e1 < 0 || l[e] > l[e1])) e1 = e;

    float w0 = 1.0f / (1.0f + __expf(l[e1] - l[e0]));
    float w1 = 1.0f - w0;

    int i0 = atomicAdd(&lcnt[e0], 1);
    int i1 = atomicAdd(&lcnt[e1], 1);
    __syncthreads();
    if (t == 0) {
        int s = 0;
        for (int e = 0; e < NUM_EXPERTS; ++e) { lbase[e] = s; s += lcnt[e]; }
    }
    __syncthreads();

    int s0 = lbase[e0] + i0;
    int s1 = lbase[e1] + i1;
    tok_of_slot[s0] = t;
    tok_of_slot[s1] = t;
    slots[2 * t] = s0;  slots[2 * t + 1] = s1;
    wts[2 * t] = w0;    wts[2 * t + 1] = w1;
    if (t < NUM_EXPERTS) { counts[t] = lcnt[t]; bases[t] = lbase[t]; }
}

// fp32 -> bf16 copy of hidden_states (4 MB, stays L2-resident, reused 8000x)
__global__ void __launch_bounds__(256)
cvt_to_bf16(const float* __restrict__ x, bf16_t* __restrict__ y, int n) {
    int i = blockIdx.x * 256 + threadIdx.x;
    if (i < n) y[i] = (bf16_t)x[i];
}

// ---------------------------------------------------------------------------
// gate_up GEMM + SwiGLU.  grid = (INTER/64, E), 256 threads = 8 waves.
// Waves 0..3: gate columns [i0+16w, +16); waves 4..7: matching up columns.
// Each wave: 4 accumulator tiles = 64 tokens x 16 cols, so every streamed
// weight fragment serves 64 tokens. K loop: HIDDEN/32 = 64 WMMAs per tile.
// ---------------------------------------------------------------------------
__global__ void __launch_bounds__(256)
moe_gateup(const bf16_t* __restrict__ X, const float* __restrict__ W13,
           const int* __restrict__ counts, const int* __restrict__ bases,
           const int* __restrict__ tok_of_slot, bf16_t* __restrict__ act) {
    const int e    = blockIdx.y;
    const int i0   = blockIdx.x * 64;
    const int wave = threadIdx.x >> 5;
    const int lane = threadIdx.x & 31;
    const int g    = lane >> 4;
    const int nl   = lane & 15;
    const bool is_up = wave >= 4;
    const int  nch   = wave & 3;
    const int  ocol  = i0 + nch * 16 + nl;   // act column / gate row index

    const float* wrow =
        W13 + ((size_t)e * (2 * INTER) + (is_up ? INTER : 0) + ocol) * HIDDEN;

    const int cnt = counts[e];
    const int sb  = bases[e];

    __shared__ float lds_gate[4][4][8][32];  // 16 KB: gate accum exchange

    for (int grp = 0; grp * 64 < cnt; ++grp) {
        const bf16_t* arow[4];
#pragma unroll
        for (int tl = 0; tl < 4; ++tl) {
            int mloc = grp * 64 + tl * 16 + nl;
            int s = sb + (mloc < cnt ? mloc : 0);      // clamp padded lanes
            arow[tl] = X + (size_t)tok_of_slot[s] * HIDDEN;
        }

        v8f acc[4] = {};
        for (int k = 0; k < HIDDEN; k += 32) {
            v16bf bf = load_b_frag_f32(wrow + k, g);
#pragma unroll
            for (int tl = 0; tl < 4; ++tl) {
                v16bf a = load_a_frag(arow[tl] + k, g);
                acc[tl] = wmma_bf16(a, bf, acc[tl]);
            }
        }

        if (!is_up) {
#pragma unroll
            for (int tl = 0; tl < 4; ++tl)
#pragma unroll
                for (int r = 0; r < 8; ++r)
                    lds_gate[wave][tl][r][lane] = acc[tl][r];
        }
        __syncthreads();
        if (is_up) {
#pragma unroll
            for (int tl = 0; tl < 4; ++tl) {
#pragma unroll
                for (int r = 0; r < 8; ++r) {
                    float gv = lds_gate[wave - 4][tl][r][lane];
                    float uv = acc[tl][r];
                    float av = (gv / (1.0f + __expf(-gv))) * uv;  // silu(g)*u
                    int mloc = grp * 64 + tl * 16 + r + 8 * g;     // D: m=r+8g
                    if (mloc < cnt)
                        act[(size_t)(sb + mloc) * INTER + ocol] = (bf16_t)av;
                }
            }
        }
        __syncthreads();
    }
}

// ---------------------------------------------------------------------------
// down GEMM.  grid = (HIDDEN/128, E), 8 waves x 16 h-cols each.
// Same 64x16-per-wave tiling; K = INTER = 4096 -> 128 WMMAs per tile.
// ---------------------------------------------------------------------------
__global__ void __launch_bounds__(256)
moe_down(const bf16_t* __restrict__ act, const float* __restrict__ W2,
         const int* __restrict__ counts, const int* __restrict__ bases,
         float* __restrict__ eo) {
    const int e    = blockIdx.y;
    const int h0   = blockIdx.x * 128;
    const int wave = threadIdx.x >> 5;
    const int lane = threadIdx.x & 31;
    const int g    = lane >> 4;
    const int nl   = lane & 15;
    const int hcol = h0 + wave * 16 + nl;

    const float* wrow = W2 + ((size_t)e * HIDDEN + hcol) * INTER;
    const int cnt = counts[e];
    const int sb  = bases[e];

    for (int grp = 0; grp * 64 < cnt; ++grp) {
        const bf16_t* arow[4];
#pragma unroll
        for (int tl = 0; tl < 4; ++tl) {
            int mloc = grp * 64 + tl * 16 + nl;
            int s = sb + (mloc < cnt ? mloc : 0);
            arow[tl] = act + (size_t)s * INTER;
        }

        v8f acc[4] = {};
        for (int k = 0; k < INTER; k += 32) {
            v16bf bf = load_b_frag_f32(wrow + k, g);
#pragma unroll
            for (int tl = 0; tl < 4; ++tl) {
                v16bf a = load_a_frag(arow[tl] + k, g);
                acc[tl] = wmma_bf16(a, bf, acc[tl]);
            }
        }

#pragma unroll
        for (int tl = 0; tl < 4; ++tl)
#pragma unroll
            for (int r = 0; r < 8; ++r) {
                int mloc = grp * 64 + tl * 16 + r + 8 * g;
                if (mloc < cnt)
                    eo[(size_t)(sb + mloc) * HIDDEN + hcol] = acc[tl][r];
            }
    }
}

// out[t,:] = w0 * eo[slot0(t),:] + w1 * eo[slot1(t),:]   (deterministic)
__global__ void __launch_bounds__(256)
moe_combine(const float* __restrict__ eo, const int* __restrict__ slots,
            const float* __restrict__ wts, float* __restrict__ out) {
    const int t = blockIdx.x;
    const int s0 = slots[2 * t], s1 = slots[2 * t + 1];
    const float w0 = wts[2 * t], w1 = wts[2 * t + 1];
    const float* r0 = eo + (size_t)s0 * HIDDEN;
    const float* r1 = eo + (size_t)s1 * HIDDEN;
    float* o = out + (size_t)t * HIDDEN;
    for (int h = threadIdx.x; h < HIDDEN; h += 256)
        o[h] = w0 * r0[h] + w1 * r1[h];
}

extern "C" void kernel_launch(void* const* d_in, const int* in_sizes, int n_in,
                              void* d_out, int out_size, void* d_ws, size_t ws_size,
                              hipStream_t stream) {
    const float* hs  = (const float*)d_in[0];   // [T, H]
    const float* rl  = (const float*)d_in[1];   // [T, E]
    const float* W13 = (const float*)d_in[2];   // [E, 2I, H]
    const float* W2  = (const float*)d_in[3];   // [E, H, I]
    // d_in[4] = top_k (fixed = 2)
    float* out = (float*)d_out;

    char* ws = (char*)d_ws;
    size_t off = 0;
    auto take = [&](size_t bytes) {
        void* p = ws + off;
        off = (off + bytes + 255) & ~(size_t)255;
        return p;
    };
    bf16_t* Xb     = (bf16_t*)take((size_t)TOKENS * HIDDEN * sizeof(bf16_t)); //  4 MB
    bf16_t* act    = (bf16_t*)take((size_t)NSLOTS * INTER * sizeof(bf16_t));  // 16 MB
    float*  eo     = (float*) take((size_t)NSLOTS * HIDDEN * sizeof(float));  // 16 MB
    int*    counts = (int*)   take(NUM_EXPERTS * sizeof(int));
    int*    bases  = (int*)   take(NUM_EXPERTS * sizeof(int));
    int*    tok    = (int*)   take(NSLOTS * sizeof(int));
    int*    slots  = (int*)   take(NSLOTS * sizeof(int));
    float*  wts    = (float*) take(NSLOTS * sizeof(float));
    (void)ws_size; (void)in_sizes; (void)n_in; (void)out_size;

    moe_router<<<1, TOKENS, 0, stream>>>(rl, counts, bases, tok, slots, wts);
    cvt_to_bf16<<<(TOKENS * HIDDEN + 255) / 256, 256, 0, stream>>>(hs, Xb, TOKENS * HIDDEN);
    moe_gateup<<<dim3(INTER / 64, NUM_EXPERTS), 256, 0, stream>>>(Xb, W13, counts, bases, tok, act);
    moe_down<<<dim3(HIDDEN / 128, NUM_EXPERTS), 256, 0, stream>>>(act, W2, counts, bases, eo);
    moe_combine<<<TOKENS, 256, 0, stream>>>(eo, slots, wts, out);
}
